// PathomicGraphNetHetero_33457795236063
// MI455X (gfx1250) — matrix operand order repaired and checked by reading, not compiled, not run on hardware
//
#include <hip/hip_runtime.h>

typedef __attribute__((ext_vector_type(16))) _Float16 v16h;
typedef __attribute__((ext_vector_type(4)))  _Float16 v4h;
typedef __attribute__((ext_vector_type(8)))  float    v8f;

static constexpr int   kDIn    = 128;
static constexpr int   kHid    = 256;
static constexpr int   kDOut   = 128;
static constexpr int   kNEdgeT = 4;
static constexpr float kBnEps  = 1e-5f;

// ---------------------------------------------------------------- WMMA core
__device__ __forceinline__ v8f wmma16x16x32(v16h a, v16h b, v8f c) {
  return __builtin_amdgcn_wmma_f32_16x16x32_f16(
      /*neg_a=*/false, a, /*neg_b=*/false, b,
      /*c_mod=*/(short)0, c, /*reuse_a=*/false, /*reuse_b=*/false);
}

// A fragment from a per-lane base pointer (row*lda + kb already folded in).
// lanes 0-15 -> M=lane&15, K in {0..7,16..23}; lanes 16-31 -> K in {8..15,24..31}.
__device__ __forceinline__ v16h load_a_at(const _Float16* __restrict__ pa) {
  v16h a;
#pragma unroll
  for (int j = 0; j < 8; ++j) {
    a[j]     = pa[j];
    a[j + 8] = pa[j + 16];
  }
  return a;
}

// 16x64 strip accumulate: software-pipelined, pointer-bumped addressing.
// B is pre-swizzled: fragment (kt,nt) = 32 lanes x 16 f16 lane-contiguous, so
// per-lane B address advances by ntiles*512 f16 per k-tile; col tiles are +512.
// NOTE: the last prefetch over-reads one k-tile past the end (never consumed);
// workspace buffers are padded to absorb it.
__device__ __forceinline__ void gemm_strip(const _Float16* __restrict__ A, int K,
                                           const _Float16* __restrict__ Bp, int ntiles,
                                           int row0, int ct0, int lane, int nrows,
                                           v8f acc[4]) {
  const int KT = K >> 5;
  int m = lane & 15;
  int row = row0 + m;
  if (row >= nrows) row = nrows - 1;   // duplicate-load only; stores are guarded
  int kb = (lane < 16) ? 0 : 8;
  const _Float16* pa = A + (size_t)row * K + kb;
  const size_t bstep = (size_t)ntiles << 9;                       // f16 per k-tile
  const _Float16* pb = Bp + (((size_t)ct0 * 32 + lane) << 4);

  v16h a0 = load_a_at(pa);
  v16h b0[4];
#pragma unroll
  for (int c = 0; c < 4; ++c) b0[c] = *(const v16h*)(pb + (c << 9));

  for (int kt = 0; kt < KT; ++kt) {
    v16h a1 = load_a_at(pa + 32);                                 // prefetch next
    v16h b1[4];
#pragma unroll
    for (int c = 0; c < 4; ++c) b1[c] = *(const v16h*)(pb + bstep + (c << 9));
#pragma unroll
    for (int c = 0; c < 4; ++c) acc[c] = wmma16x16x32(a0, b0[c], acc[c]);
    pa += 32;
    pb += bstep;
    a0 = a1;
#pragma unroll
    for (int c = 0; c < 4; ++c) b0[c] = b1[c];
  }
}

// ---------------------------------------------------------------- utilities
__global__ void k_f32_to_f16(const float* __restrict__ in, _Float16* __restrict__ out, long n) {
  long i = (long)blockIdx.x * blockDim.x + threadIdx.x;
  if (i < n) out[i] = (_Float16)in[i];
}

__global__ void k_zero_f32(float* __restrict__ p, long n) {
  long i = (long)blockIdx.x * blockDim.x + threadIdx.x;
  if (i < n) p[i] = 0.0f;
}

// f32 row-major W[K][Ncols] -> WMMA-B-fragment-packed f16.
__global__ void k_pack_b_f16(const float* __restrict__ W, _Float16* __restrict__ P,
                             int K, int Ncols) {
  int i = blockIdx.x * blockDim.x + threadIdx.x;
  int total = K * Ncols;
  if (i >= total) return;
  int e    = i & 15;
  int l    = (i >> 4) & 31;
  int tile = i >> 9;
  int ntiles = Ncols >> 4;
  int nt = tile % ntiles;
  int kt = tile / ntiles;
  int n  = (nt << 4) + (l & 15);
  int k  = (kt << 5) + ((l < 16) ? 0 : 8) + (e & 7) + ((e & 8) << 1);
  P[i] = (_Float16)W[(size_t)k * Ncols + n];
}

// ---------------------------------------------------------------- GEMM kernels
// Multi-wave blocks: wave w handles col-strip (blockIdx.y*nwaves + w)*64.
__device__ __forceinline__ void strip_coords(int& row0, int& ct0, int& lane) {
  row0 = blockIdx.x * 16;
  int wave = threadIdx.x >> 5;
  lane = threadIdx.x & 31;
  ct0 = (blockIdx.y * (blockDim.x >> 5) + wave) * 4;
}

// C (f16) = A @ B + bias
__global__ void k_gemm_bias_f16(const _Float16* __restrict__ A, const _Float16* __restrict__ Bp,
                                const float* __restrict__ bias, _Float16* __restrict__ C,
                                int nrows, int K, int ncols) {
  int row0, ct0, lane;
  strip_coords(row0, ct0, lane);
  v8f acc[4] = {v8f{}, v8f{}, v8f{}, v8f{}};
  gemm_strip(A, K, Bp, ncols >> 4, row0, ct0, lane, nrows, acc);
  int mb = (lane < 16) ? 0 : 8;
  bool full = (row0 + 16 <= nrows);
#pragma unroll
  for (int c = 0; c < 4; ++c) {
    int n = ((ct0 + c) << 4) + (lane & 15);
    float bv = bias[n];
    if (full) {
#pragma unroll
      for (int r = 0; r < 8; ++r)
        C[(size_t)(row0 + mb + r) * ncols + n] = (_Float16)(acc[c][r] + bv);
    } else {
#pragma unroll
      for (int r = 0; r < 8; ++r) {
        int row = row0 + mb + r;
        if (row < nrows) C[(size_t)row * ncols + n] = (_Float16)(acc[c][r] + bv);
      }
    }
  }
}

// H (f16) = relu(BN(A @ B))
__global__ void k_gemm_bn_relu_f16(const _Float16* __restrict__ A, const _Float16* __restrict__ Bp,
                                   const float* __restrict__ gamma, const float* __restrict__ beta,
                                   const float* __restrict__ mean, const float* __restrict__ var,
                                   _Float16* __restrict__ C, int nrows, int K, int ncols) {
  int row0, ct0, lane;
  strip_coords(row0, ct0, lane);
  v8f acc[4] = {v8f{}, v8f{}, v8f{}, v8f{}};
  gemm_strip(A, K, Bp, ncols >> 4, row0, ct0, lane, nrows, acc);
  int mb = (lane < 16) ? 0 : 8;
  bool full = (row0 + 16 <= nrows);
#pragma unroll
  for (int c = 0; c < 4; ++c) {
    int n = ((ct0 + c) << 4) + (lane & 15);
    float sc = gamma[n] * rsqrtf(var[n] + kBnEps);
    float sh = beta[n] - mean[n] * sc;
    if (full) {
#pragma unroll
      for (int r = 0; r < 8; ++r)
        C[(size_t)(row0 + mb + r) * ncols + n] = (_Float16)fmaxf(acc[c][r] * sc + sh, 0.0f);
    } else {
#pragma unroll
      for (int r = 0; r < 8; ++r) {
        int row = row0 + mb + r;
        if (row < nrows)
          C[(size_t)row * ncols + n] = (_Float16)fmaxf(acc[c][r] * sc + sh, 0.0f);
      }
    }
  }
}

// Hall[n][t][128] (f16) = H[n] @ W_hetero[t]  (grid.z = edge type)
__global__ void k_gemm_hetero_f16(const _Float16* __restrict__ H, const _Float16* __restrict__ Whp,
                                  _Float16* __restrict__ Hall, int nrows) {
  int row0, ct0, lane;
  strip_coords(row0, ct0, lane);
  int t = blockIdx.z;
  const _Float16* Bp = Whp + (size_t)t * kDIn * kDOut;
  v8f acc[4] = {v8f{}, v8f{}, v8f{}, v8f{}};
  gemm_strip(H, kDIn, Bp, kDOut >> 4, row0, ct0, lane, nrows, acc);
  int mb = (lane < 16) ? 0 : 8;
  bool full = (row0 + 16 <= nrows);
#pragma unroll
  for (int c = 0; c < 4; ++c) {
    int n = ((ct0 + c) << 4) + (lane & 15);
    if (full) {
#pragma unroll
      for (int r = 0; r < 8; ++r)
        Hall[((size_t)(row0 + mb + r) * kNEdgeT + t) * kDOut + n] = (_Float16)acc[c][r];
    } else {
#pragma unroll
      for (int r = 0; r < 8; ++r) {
        int row = row0 + mb + r;
        if (row < nrows)
          Hall[((size_t)row * kNEdgeT + t) * kDOut + n] = (_Float16)acc[c][r];
      }
    }
  }
}

// Edge stage: one wave per edge; L2-resident f16 gather, f32 atomic scatter.
__global__ void k_aggregate(const _Float16* __restrict__ Hall, const int* __restrict__ src,
                            const int* __restrict__ dst, const int* __restrict__ etype,
                            float* __restrict__ nsum, float* __restrict__ deg, int E) {
  int wid  = (int)(((long)blockIdx.x * blockDim.x + threadIdx.x) >> 5);
  int lane = threadIdx.x & 31;
  if (wid >= E) return;
  int s = src[wid], d = dst[wid], t = etype[wid];
  const _Float16* row = Hall + ((size_t)s * kNEdgeT + t) * kDOut;
  float* orow = nsum + (size_t)d * kDOut + lane * 4;
  v4h v = *(const v4h*)(row + lane * 4);        // one global_load_b64 per lane
#pragma unroll
  for (int j = 0; j < 4; ++j) atomicAdd(&orow[j], (float)v[j]);
  if (lane == 0) atomicAdd(&deg[d], 1.0f);
}

// Self path for BOTH node types + final combine.
__global__ void k_gemm_self_combine(const _Float16* __restrict__ S, const _Float16* __restrict__ Wsp,
                                    const int* __restrict__ ntype, const float* __restrict__ bias,
                                    const float* __restrict__ nsum, const float* __restrict__ deg,
                                    float* __restrict__ out, int nrows) {
  int row0, ct0, lane;
  strip_coords(row0, ct0, lane);
  const int ntiles = kDOut >> 4;
  v8f acc0[4] = {v8f{}, v8f{}, v8f{}, v8f{}};
  v8f acc1[4] = {v8f{}, v8f{}, v8f{}, v8f{}};
  gemm_strip(S, kHid, Wsp, ntiles, row0, ct0, lane, nrows, acc0);
  gemm_strip(S, kHid, Wsp + (size_t)kHid * kDOut, ntiles, row0, ct0, lane, nrows, acc1);
  int mb = (lane < 16) ? 0 : 8;
#pragma unroll
  for (int c = 0; c < 4; ++c) {
    int n = ((ct0 + c) << 4) + (lane & 15);
    float bv = bias[n];
#pragma unroll
    for (int r = 0; r < 8; ++r) {
      int row = row0 + mb + r;
      if (row < nrows) {
        float sv = (ntype[row] == 0) ? acc0[c][r] : acc1[c][r];
        float dg = deg[row]; dg = dg > 1.0f ? dg : 1.0f;
        out[(size_t)row * kDOut + n] = sv + bv + nsum[(size_t)row * kDOut + n] / dg;
      }
    }
  }
}

// ---------------------------------------------------------------- launch
extern "C" void kernel_launch(void* const* d_in, const int* in_sizes, int n_in,
                              void* d_out, int out_size, void* d_ws, size_t ws_size,
                              hipStream_t stream) {
  const float* feat  = (const float*)d_in[0];
  const float* W1    = (const float*)d_in[1];
  const float* b1    = (const float*)d_in[2];
  const float* W2    = (const float*)d_in[3];
  const float* gam   = (const float*)d_in[4];
  const float* bet   = (const float*)d_in[5];
  const float* mu    = (const float*)d_in[6];
  const float* var   = (const float*)d_in[7];
  const float* Wh    = (const float*)d_in[8];
  const float* Ws    = (const float*)d_in[9];
  const float* bs    = (const float*)d_in[10];
  const float* Wself = (const float*)d_in[11];
  const float* bias  = (const float*)d_in[12];
  const int*   src   = (const int*)d_in[13];
  const int*   dst   = (const int*)d_in[14];
  const int*   etype = (const int*)d_in[15];
  const int*   ntype = (const int*)d_in[16];
  float* out = (float*)d_out;

  const int N = in_sizes[0] / kDIn;
  const int E = in_sizes[13];

  // Workspace carve-out; every buffer gets 8KB tail padding to absorb the
  // branch-free prefetch over-read of the pipelined GEMM loop.
  char* p = (char*)d_ws;
  auto take = [&](size_t bytes) -> void* {
    void* r = (void*)p;
    p += (bytes + 8192 + 255) & ~(size_t)255;
    return r;
  };
  _Float16* feat16 = (_Float16*)take((size_t)N * kDIn * 2);
  _Float16* W1p    = (_Float16*)take((size_t)kDIn * kHid * 2);
  _Float16* W2p    = (_Float16*)take((size_t)kHid * kDIn * 2);
  _Float16* Whp    = (_Float16*)take((size_t)kNEdgeT * kDIn * kDOut * 2);
  _Float16* Wsp    = (_Float16*)take((size_t)kDIn * kHid * 2);
  _Float16* Wselfp = (_Float16*)take((size_t)2 * kHid * kDOut * 2);
  _Float16* tmp1   = (_Float16*)take((size_t)N * kHid * 2);            // feat@W1+b1
  _Float16* h16    = (_Float16*)take((size_t)N * kDIn * 2);            // relu(bn(.@W2))
  _Float16* hall   = (_Float16*)take((size_t)N * kNEdgeT * kDOut * 2); // 102 MB -> L2
  _Float16* s16    = (_Float16*)take((size_t)N * kHid * 2);            // feat@Ws+bs
  float*    nsum   = (float*)take((size_t)N * kDOut * 4);
  float*    deg    = (float*)take((size_t)N * 4);

  {
    long n = (long)N * kDIn;
    k_f32_to_f16<<<dim3((unsigned)((n + 255) / 256)), dim3(256), 0, stream>>>(feat, feat16, n);
  }
  auto pack = [&](const float* W, _Float16* P, int K, int Nc) {
    int total = K * Nc;
    k_pack_b_f16<<<dim3((total + 255) / 256), dim3(256), 0, stream>>>(W, P, K, Nc);
  };
  pack(W1, W1p, kDIn, kHid);
  pack(W2, W2p, kHid, kDIn);
  for (int t = 0; t < kNEdgeT; ++t)
    pack(Wh + (size_t)t * kDIn * kDOut, Whp + (size_t)t * kDIn * kDOut, kDIn, kDOut);
  pack(Ws, Wsp, kDIn, kHid);
  for (int t = 0; t < 2; ++t)
    pack(Wself + (size_t)t * kHid * kDOut, Wselfp + (size_t)t * kHid * kDOut, kHid, kDOut);

  k_zero_f32<<<dim3((unsigned)(((long)N * kDOut + 255) / 256)), dim3(256), 0, stream>>>(
      nsum, (long)N * kDOut);
  k_zero_f32<<<dim3((unsigned)((N + 255) / 256)), dim3(256), 0, stream>>>(deg, (long)N);

  const int RT = (N + 15) / 16;

  // fc_neigh: Linear(128->256)+b  ->  Linear(256->128) + BN + ReLU
  k_gemm_bias_f16<<<dim3(RT, 1), dim3(128), 0, stream>>>(     // 4 waves x 64 cols = 256
      feat16, W1p, b1, tmp1, N, kDIn, kHid);
  k_gemm_bn_relu_f16<<<dim3(RT, 1), dim3(64), 0, stream>>>(   // 2 waves x 64 cols = 128
      tmp1, W2p, gam, bet, mu, var, h16, N, kHid, kDIn);

  // hetero typed linear: h_all[n][t] = h[n] @ W_hetero[t]
  k_gemm_hetero_f16<<<dim3(RT, 1, kNEdgeT), dim3(64), 0, stream>>>(
      h16, Whp, hall, N);

  // edge gather + mean-sum scatter
  k_aggregate<<<dim3((unsigned)((E + 7) / 8)), dim3(256), 0, stream>>>(
      hall, src, dst, etype, nsum, deg, E);

  // self path: s = feat@Ws+bs ; out = fc_self(s, node_type) + bias + mean
  k_gemm_bias_f16<<<dim3(RT, 1), dim3(128), 0, stream>>>(
      feat16, Wsp, bs, s16, N, kDIn, kHid);
  k_gemm_self_combine<<<dim3(RT, 1), dim3(64), 0, stream>>>(
      s16, Wselfp, ntype, bias, nsum, deg, out, N);
}
